// SiameseGCN_83597243449445
// MI455X (gfx1250) — matrix operand order repaired and checked by reading, not compiled, not run on hardware
//
#include <hip/hip_runtime.h>
#include <hip/hip_bf16.h>
#include <math.h>

// ---------------- problem constants (match reference) ----------------
#define NN      50000     // nodes
#define NE      600000    // edges
#define DIM     128
#define N_ET    4
#define STEPS   5
#define HEADS   2
#define NG      64
#define NC      32

typedef __attribute__((ext_vector_type(16))) __bf16 v16bf;
typedef __attribute__((ext_vector_type(8)))  __bf16 v8bf;
typedef __attribute__((ext_vector_type(8)))  float  v8f;

// ====================================================================
// WMMA GEMM:  C[M x Nout] = A_bf16[M x 128] * W_bf16[Nout x 128]^T + bias
// One wave per 16x16 output tile; K=128 unrolled into 4 wmma ops.
// A fragment (16-bit A 16x32): lane&15 = row M,
//   half i -> K = ((lane>>4)<<4) + i   (contiguous 16 halves per lane)
// B fragment (16-bit B 32x16): lane&15 = col N,
//   half i -> K = ((lane>>4)<<4) + i   (contiguous 16 halves per lane)
// C/D f32: lane&15 = N, vgpr r -> M = r + 8*(lane>>4)
// ====================================================================
__global__ __launch_bounds__(256) void gemm_bf16_k128(
    const __bf16* __restrict__ A, const __bf16* __restrict__ W,
    const float* __restrict__ bias, float* __restrict__ C, int M, int Nout)
{
    const int lane  = threadIdx.x & 31;
    const int wave  = threadIdx.x >> 5;
    const int tileM = blockIdx.x * 8 + wave;
    if (tileM * 16 >= M) return;
    const int col  = blockIdx.y * 16 + (lane & 15);
    const int rowA = tileM * 16 + (lane & 15);

    v8f acc;
    const float bv = bias ? bias[col] : 0.0f;
#pragma unroll
    for (int r = 0; r < 8; ++r) acc[r] = bv;

    const __bf16* pA = A + (size_t)rowA * DIM + ((lane >> 4) << 4);
    const __bf16* pB = W + (size_t)col  * DIM + ((lane >> 4) << 4);

#pragma unroll
    for (int k0 = 0; k0 < DIM; k0 += 32) {
        v16bf af = *(const v16bf*)(pA + k0);
        v16bf bf = *(const v16bf*)(pB + k0);
        acc = __builtin_amdgcn_wmma_f32_16x16x32_bf16(
                  false, af, false, bf, (short)0, acc, false, false);
    }

    const int rowBase = tileM * 16 + ((lane >> 4) << 3);
#pragma unroll
    for (int r = 0; r < 8; ++r)
        C[(size_t)(rowBase + r) * Nout + col] = acc[r];
}

// ---------------- utility kernels ----------------
__global__ void fill_f32_k(float* __restrict__ p, float v, size_t n) {
    size_t i = (size_t)blockIdx.x * blockDim.x + threadIdx.x;
    if (i < n) p[i] = v;
}
__global__ void fill_u32_k(unsigned* __restrict__ p, unsigned v, size_t n) {
    size_t i = (size_t)blockIdx.x * blockDim.x + threadIdx.x;
    if (i < n) p[i] = v;
}
__global__ void cvt_bf16_k(const float* __restrict__ in, __bf16* __restrict__ out, size_t n) {
    size_t i = (size_t)blockIdx.x * blockDim.x + threadIdx.x;
    if (i < n) out[i] = (__bf16)in[i];
}
__global__ void copy_cvt_k(const float* __restrict__ in, float* __restrict__ out,
                           __bf16* __restrict__ outb, size_t n) {
    size_t i = (size_t)blockIdx.x * blockDim.x + threadIdx.x;
    if (i < n) { float v = in[i]; out[i] = v; outb[i] = (__bf16)v; }
}

// ---------------- message scatter: a[dst] += t[etype, src]  (wave/edge) ----------------
__global__ __launch_bounds__(256) void edge_scatter_k(
    const float* __restrict__ t, const int* __restrict__ src,
    const int* __restrict__ dst, const int* __restrict__ et,
    float* __restrict__ a, int E)
{
    int w    = (blockIdx.x * blockDim.x + threadIdx.x) >> 5;
    int lane = threadIdx.x & 31;
    if (w >= E) return;
    int s = src[w], d = dst[w], k = et[w];
    const float4 v = *(const float4*)(t + ((size_t)k * NN + s) * DIM + lane * 4);
    float* pa = a + (size_t)d * DIM + lane * 4;
    atomicAdd(pa + 0, v.x); atomicAdd(pa + 1, v.y);
    atomicAdd(pa + 2, v.z); atomicAdd(pa + 3, v.w);
}

// ---------------- fused GRU gates; also emits bf16 h for next step ----------------
__device__ __forceinline__ float sigm(float x) { return 1.0f / (1.0f + __expf(-x)); }

__global__ void gru_gate_k(const float* __restrict__ gi, const float* __restrict__ gh,
                           float* __restrict__ h, __bf16* __restrict__ hb)
{
    size_t idx = (size_t)blockIdx.x * blockDim.x + threadIdx.x;
    if (idx >= (size_t)NN * DIM) return;
    size_t n = idx >> 7; int d = (int)(idx & 127);
    const float* gin = gi + n * 3 * DIM;
    const float* ghn = gh + n * 3 * DIM;
    float r  = sigm(gin[d] + ghn[d]);
    float z  = sigm(gin[DIM + d] + ghn[DIM + d]);
    float nn = tanhf(gin[2 * DIM + d] + r * ghn[2 * DIM + d]);
    float hv = (1.0f - z) * nn + z * h[idx];
    h[idx] = hv; hb[idx] = (__bf16)hv;
}

// ---------------- L2-normalize rows then sigmoid (wave per node) ----------------
__global__ void norm_sigmoid_k(float* __restrict__ h, __bf16* __restrict__ hb)
{
    int n    = (blockIdx.x * blockDim.x + threadIdx.x) >> 5;
    int lane = threadIdx.x & 31;
    if (n >= NN) return;
    float* ph = h + (size_t)n * DIM + lane * 4;
    float4 v = *(const float4*)ph;
    float ss = v.x*v.x + v.y*v.y + v.z*v.z + v.w*v.w;
    for (int off = 16; off; off >>= 1) ss += __shfl_xor(ss, off, 32);
    float inv = 1.0f / fmaxf(sqrtf(ss), 1e-12f);
    float4 o; o.x = sigm(v.x*inv); o.y = sigm(v.y*inv); o.z = sigm(v.z*inv); o.w = sigm(v.w*inv);
    *(float4*)ph = o;
    __bf16* pb = hb + (size_t)n * DIM + lane * 4;
    pb[0] = (__bf16)o.x; pb[1] = (__bf16)o.y; pb[2] = (__bf16)o.z; pb[3] = (__bf16)o.w;
}

// ---------------- el/er attention scores (wave per node, both heads) ----------------
__global__ void attn_scores_k(const float* __restrict__ zft,
                              const float* __restrict__ al, const float* __restrict__ ar,
                              float* __restrict__ el, float* __restrict__ er)
{
    int n    = (blockIdx.x * blockDim.x + threadIdx.x) >> 5;
    int lane = threadIdx.x & 31;
    if (n >= NN) return;
    float s[4] = {0.f, 0.f, 0.f, 0.f};          // l0, r0, l1, r1
#pragma unroll
    for (int hd = 0; hd < HEADS; ++hd) {
#pragma unroll
        for (int j = 0; j < 4; ++j) {
            int d = lane * 4 + j;
            float z = zft[((size_t)n * HEADS + hd) * DIM + d];
            s[2*hd+0] += z * al[hd * DIM + d];
            s[2*hd+1] += z * ar[hd * DIM + d];
        }
    }
#pragma unroll
    for (int j = 0; j < 4; ++j)
        for (int off = 16; off; off >>= 1) s[j] += __shfl_xor(s[j], off, 32);
    if (lane == 0) {
        el[n*HEADS+0] = s[0]; er[n*HEADS+0] = s[1];
        el[n*HEADS+1] = s[2]; er[n*HEADS+1] = s[3];
    }
}

// ordered-uint encoding for float atomicMax (handles negatives)
__device__ __forceinline__ unsigned enc_f32(float f) {
    unsigned u = __float_as_uint(f);
    return (u & 0x80000000u) ? ~u : (u | 0x80000000u);
}
__device__ __forceinline__ float dec_f32(unsigned u) {
    return __uint_as_float((u & 0x80000000u) ? (u & 0x7FFFFFFFu) : ~u);
}

__global__ void edge_max_k(const float* __restrict__ el, const float* __restrict__ er,
                           const int* __restrict__ src, const int* __restrict__ dst,
                           float* __restrict__ ebuf, unsigned* __restrict__ menc)
{
    int i = blockIdx.x * blockDim.x + threadIdx.x;
    if (i >= NE * HEADS) return;
    int e = i >> 1, hd = i & 1;
    float v = el[src[e]*HEADS+hd] + er[dst[e]*HEADS+hd];
    v = (v > 0.0f) ? v : 0.2f * v;                 // leaky_relu 0.2
    ebuf[i] = v;
    atomicMax(menc + dst[e]*HEADS+hd, enc_f32(v));
}

__global__ void edge_expsum_k(float* __restrict__ ebuf, const unsigned* __restrict__ menc,
                              const int* __restrict__ dst, float* __restrict__ den)
{
    int i = blockIdx.x * blockDim.x + threadIdx.x;
    if (i >= NE * HEADS) return;
    int e = i >> 1, hd = i & 1;
    float m  = dec_f32(menc[dst[e]*HEADS+hd]);
    float ex = __expf(ebuf[i] - m);
    ebuf[i] = ex;
    atomicAdd(den + dst[e]*HEADS+hd, ex);
}

// rst[dst,hd,:] += (ex/den[dst,hd]) * zft[src,hd,:]    (wave per (edge,head))
__global__ __launch_bounds__(256) void edge_apply_k(
    const float* __restrict__ ebuf, const float* __restrict__ den,
    const float* __restrict__ zft, const int* __restrict__ src,
    const int* __restrict__ dst, float* __restrict__ rst)
{
    int w    = (blockIdx.x * blockDim.x + threadIdx.x) >> 5;
    int lane = threadIdx.x & 31;
    if (w >= NE * HEADS) return;
    int e = w >> 1, hd = w & 1;
    int d = dst[e];
    float coef = ebuf[w] / den[d*HEADS+hd];
    const float4 z = *(const float4*)(zft + ((size_t)src[e]*HEADS + hd)*DIM + lane*4);
    float* pr = rst + ((size_t)d*HEADS + hd)*DIM + lane*4;
    atomicAdd(pr + 0, coef * z.x); atomicAdd(pr + 1, coef * z.y);
    atomicAdd(pr + 2, coef * z.z); atomicAdd(pr + 3, coef * z.w);
}

// relu(rst + bias) then per-graph sum pool + node count
__global__ void node_pool_k(const float* __restrict__ rst, const float* __restrict__ gb,
                            const int* __restrict__ gid, float* __restrict__ sums,
                            float* __restrict__ cnt)
{
    size_t idx = (size_t)blockIdx.x * blockDim.x + threadIdx.x;
    if (idx >= (size_t)NN * HEADS * DIM) return;
    size_t n = idx >> 8; int j = (int)(idx & 255);
    float v = fmaxf(rst[idx] + gb[j], 0.0f);
    int g = gid[n];
    atomicAdd(sums + (size_t)g * HEADS * DIM + j, v);
    if (j == 0) atomicAdd(cnt + g, 1.0f);
}

// mean pool + tiny classifier (block per (graph, head))
__global__ void classify_k(const float* __restrict__ sums, const float* __restrict__ cnt,
                           const float* __restrict__ Wc, const float* __restrict__ bc,
                           float* __restrict__ logits)
{
    __shared__ float hg[DIM];
    int g = blockIdx.x, hd = blockIdx.y, tid = threadIdx.x;
    float c = fmaxf(cnt[g], 1.0f);
    hg[tid] = sums[((size_t)g * HEADS + hd) * DIM + tid] / c;
    __syncthreads();
    if (tid < NC) {
        float acc = bc[tid];
#pragma unroll 4
        for (int d = 0; d < DIM; ++d) acc += hg[d] * Wc[tid * DIM + d];
        logits[((size_t)g * HEADS + hd) * NC + tid] = acc;
    }
}

// pairwise L2 over classes + softmax over the 2 heads
__global__ void pairwise_k(const float* __restrict__ l1, const float* __restrict__ l2,
                           float* __restrict__ out)
{
    int g = blockIdx.x * blockDim.x + threadIdx.x;
    if (g >= NG) return;
    float d[HEADS];
#pragma unroll
    for (int hd = 0; hd < HEADS; ++hd) {
        float ss = 0.0f;
        for (int c = 0; c < NC; ++c) {
            float t = l1[((size_t)g*HEADS+hd)*NC + c] - l2[((size_t)g*HEADS+hd)*NC + c] + 1e-6f;
            ss += t * t;
        }
        d[hd] = sqrtf(ss);
    }
    float mx = fmaxf(d[0], d[1]);
    float e0 = __expf(d[0] - mx), e1 = __expf(d[1] - mx);
    float inv = 1.0f / (e0 + e1);
    out[g*HEADS+0] = e0 * inv;
    out[g*HEADS+1] = e1 * inv;
}

// ====================================================================
// host side
// ====================================================================
static inline size_t a256(size_t x) { return (x + 255) & ~(size_t)255; }
static inline int dg(size_t n, int b) { return (int)((n + b - 1) / b); }

static void run_ggann(const float* feat, const int* src, const int* dst,
                      const int* et, const int* gid,
                      const __bf16* wb_et, const __bf16* wb_ih, const __bf16* wb_hh,
                      const __bf16* wb_fc,
                      const float* b_et, const float* b_ih, const float* b_hh,
                      const float* attn_l, const float* attn_r, const float* gat_bias,
                      const float* W_cls, const float* b_cls,
                      float* h, __bf16* hb, float* a, __bf16* ab,
                      float* t, float* gi, float* gh,
                      float* logits, hipStream_t stream)
{
    const int B = 256;
    const size_t ND = (size_t)NN * DIM;
    const int mtiles = (NN / 16 + 7) / 8;        // 50000/16 = 3125 tiles exactly

    copy_cvt_k<<<dg(ND, B), B, 0, stream>>>(feat, h, hb, ND);

    for (int s = 0; s < STEPS; ++s) {
        for (int k = 0; k < N_ET; ++k)
            gemm_bf16_k128<<<dim3(mtiles, DIM/16), B, 0, stream>>>(
                hb, wb_et + (size_t)k*DIM*DIM, b_et + k*DIM,
                t + (size_t)k*ND, NN, DIM);
        fill_f32_k<<<dg(ND, B), B, 0, stream>>>(a, 0.0f, ND);
        edge_scatter_k<<<dg((size_t)NE*32, B), B, 0, stream>>>(t, src, dst, et, a, NE);
        cvt_bf16_k<<<dg(ND, B), B, 0, stream>>>(a, ab, ND);
        gemm_bf16_k128<<<dim3(mtiles, 3*DIM/16), B, 0, stream>>>(ab, wb_ih, b_ih, gi, NN, 3*DIM);
        gemm_bf16_k128<<<dim3(mtiles, 3*DIM/16), B, 0, stream>>>(hb, wb_hh, b_hh, gh, NN, 3*DIM);
        gru_gate_k<<<dg(ND, B), B, 0, stream>>>(gi, gh, h, hb);
    }

    norm_sigmoid_k<<<dg((size_t)NN*32, B), B, 0, stream>>>(h, hb);

    // ---- GAT: alias freed buffers ----
    float* zft = t;                                   // NN*256 f32
    float* rst = t + (size_t)NN * HEADS * DIM;        // NN*256 f32
    float* el  = gi;
    float* er  = el + (size_t)NN * HEADS;
    float* eb  = er + (size_t)NN * HEADS;             // NE*2
    float* den = eb + (size_t)NE * HEADS;             // NN*2
    unsigned* menc = (unsigned*)(den + (size_t)NN * HEADS);  // NN*2
    float* sums = (float*)(menc + (size_t)NN * HEADS);       // NG*256
    float* cnt  = sums + (size_t)NG * HEADS * DIM;           // NG

    gemm_bf16_k128<<<dim3(mtiles, HEADS*DIM/16), B, 0, stream>>>(
        hb, wb_fc, (const float*)nullptr, zft, NN, HEADS*DIM);
    attn_scores_k<<<dg((size_t)NN*32, B), B, 0, stream>>>(zft, attn_l, attn_r, el, er);

    fill_u32_k<<<dg((size_t)NN*HEADS, B), B, 0, stream>>>(menc, 0x007FFFFFu, (size_t)NN*HEADS); // enc(-inf)
    fill_f32_k<<<dg((size_t)NN*HEADS, B), B, 0, stream>>>(den, 0.0f, (size_t)NN*HEADS);
    fill_f32_k<<<dg((size_t)NN*HEADS*DIM, B), B, 0, stream>>>(rst, 0.0f, (size_t)NN*HEADS*DIM);
    fill_f32_k<<<dg((size_t)NG*HEADS*DIM, B), B, 0, stream>>>(sums, 0.0f, (size_t)NG*HEADS*DIM);
    fill_f32_k<<<dg((size_t)NG, B), B, 0, stream>>>(cnt, 0.0f, (size_t)NG);

    edge_max_k<<<dg((size_t)NE*HEADS, B), B, 0, stream>>>(el, er, src, dst, eb, menc);
    edge_expsum_k<<<dg((size_t)NE*HEADS, B), B, 0, stream>>>(eb, menc, dst, den);
    edge_apply_k<<<dg((size_t)NE*HEADS*32, B), B, 0, stream>>>(eb, den, zft, src, dst, rst);
    node_pool_k<<<dg((size_t)NN*HEADS*DIM, B), B, 0, stream>>>(rst, gat_bias, gid, sums, cnt);
    classify_k<<<dim3(NG, HEADS), DIM, 0, stream>>>(sums, cnt, W_cls, b_cls, logits);
}

extern "C" void kernel_launch(void* const* d_in, const int* in_sizes, int n_in,
                              void* d_out, int out_size, void* d_ws, size_t ws_size,
                              hipStream_t stream)
{
    (void)in_sizes; (void)n_in; (void)out_size; (void)ws_size;

    const float* in_feat1 = (const float*)d_in[0];
    const float* in_feat2 = (const float*)d_in[1];
    const int*   src1 = (const int*)d_in[2];
    const int*   dst1 = (const int*)d_in[3];
    const int*   et1  = (const int*)d_in[4];
    const int*   gid1 = (const int*)d_in[5];
    const int*   src2 = (const int*)d_in[6];
    const int*   dst2 = (const int*)d_in[7];
    const int*   et2  = (const int*)d_in[8];
    const int*   gid2 = (const int*)d_in[9];
    const float* W_et = (const float*)d_in[10];
    const float* b_et = (const float*)d_in[11];
    const float* W_ih = (const float*)d_in[12];
    const float* W_hh = (const float*)d_in[13];
    const float* b_ih = (const float*)d_in[14];
    const float* b_hh = (const float*)d_in[15];
    const float* W_fc = (const float*)d_in[16];
    const float* attn_l = (const float*)d_in[17];
    const float* attn_r = (const float*)d_in[18];
    const float* gat_bias = (const float*)d_in[19];
    const float* W_cls = (const float*)d_in[20];
    const float* b_cls = (const float*)d_in[21];

    // ---- bump allocation from workspace ----
    char* p = (char*)d_ws;
    auto alloc = [&](size_t bytes) -> char* { char* r = p; p += a256(bytes); return r; };
    const size_t ND = (size_t)NN * DIM;

    __bf16* wb_et = (__bf16*)alloc((size_t)N_ET*DIM*DIM * sizeof(__bf16));
    __bf16* wb_ih = (__bf16*)alloc((size_t)3*DIM*DIM * sizeof(__bf16));
    __bf16* wb_hh = (__bf16*)alloc((size_t)3*DIM*DIM * sizeof(__bf16));
    __bf16* wb_fc = (__bf16*)alloc((size_t)HEADS*DIM*DIM * sizeof(__bf16));
    float*  logits1 = (float*)alloc((size_t)NG*HEADS*NC * sizeof(float));
    float*  logits2 = (float*)alloc((size_t)NG*HEADS*NC * sizeof(float));
    float*  h  = (float*)alloc(ND * sizeof(float));
    __bf16* hb = (__bf16*)alloc(ND * sizeof(__bf16));
    float*  a  = (float*)alloc(ND * sizeof(float));
    __bf16* ab = (__bf16*)alloc(ND * sizeof(__bf16));
    float*  t  = (float*)alloc((size_t)N_ET * ND * sizeof(float));   // also hosts zft/rst later
    float*  gi = (float*)alloc((size_t)NN * 3 * DIM * sizeof(float)); // also hosts GAT small bufs
    float*  gh = (float*)alloc((size_t)NN * 3 * DIM * sizeof(float));

    const int B = 256;
    cvt_bf16_k<<<dg((size_t)N_ET*DIM*DIM, B), B, 0, stream>>>(W_et, wb_et, (size_t)N_ET*DIM*DIM);
    cvt_bf16_k<<<dg((size_t)3*DIM*DIM, B), B, 0, stream>>>(W_ih, wb_ih, (size_t)3*DIM*DIM);
    cvt_bf16_k<<<dg((size_t)3*DIM*DIM, B), B, 0, stream>>>(W_hh, wb_hh, (size_t)3*DIM*DIM);
    cvt_bf16_k<<<dg((size_t)HEADS*DIM*DIM, B), B, 0, stream>>>(W_fc, wb_fc, (size_t)HEADS*DIM*DIM);

    run_ggann(in_feat1, src1, dst1, et1, gid1, wb_et, wb_ih, wb_hh, wb_fc,
              b_et, b_ih, b_hh, attn_l, attn_r, gat_bias, W_cls, b_cls,
              h, hb, a, ab, t, gi, gh, logits1, stream);
    run_ggann(in_feat2, src2, dst2, et2, gid2, wb_et, wb_ih, wb_hh, wb_fc,
              b_et, b_ih, b_hh, attn_l, attn_r, gat_bias, W_cls, b_cls,
              h, hb, a, ab, t, gi, gh, logits2, stream);

    pairwise_k<<<1, NG, 0, stream>>>(logits1, logits2, (float*)d_out);
}
